// RWKV6Block_75496935129739
// MI455X (gfx1250) — matrix operand order repaired and checked by reading, not compiled, not run on hardware
//
#include <hip/hip_runtime.h>
#include <hip/hip_bf16.h>
#include <math.h>

// ---------------- problem constants (RWKV6 block) ----------------
#define kB   8
#define kT   2048
#define kC   1024
#define kH   16
#define kN   64
#define kF   3584
#define kTM  32
#define kTD  64
#define kBT  (kB * kT)
#define kBTC ((size_t)kBT * kC)

// ---------------- WMMA types (CDNA5 / gfx1250, wave32) ----------------
typedef __attribute__((ext_vector_type(16))) __bf16 v16bf;
typedef __attribute__((ext_vector_type(8)))  __bf16 bf16x8;
typedef __attribute__((ext_vector_type(8)))  float  v8f;

__device__ __forceinline__ __bf16 f2bf(float f) {
  union { float f; unsigned u; } a; a.f = f;
  unsigned r = a.u + 0x7FFFu + ((a.u >> 16) & 1u);   // round-to-nearest-even
  unsigned short h = (unsigned short)(r >> 16);
  __bf16 o; __builtin_memcpy(&o, &h, 2);
  return o;
}

__device__ __forceinline__ float sigm_(float x) { return 1.f / (1.f + __expf(-x)); }

// ------- fp32[K][N] -> bf16[N][K] tiled transpose (weights become GEMM-B) -------
__global__ __launch_bounds__(256) void cvt_t_f32_bf16(const float* __restrict__ in,
                                                      __bf16* __restrict__ out,
                                                      int K, int N) {
  __shared__ float tile[32][33];
  const int kb = blockIdx.y * 32, nb = blockIdx.x * 32;
  const int tx = threadIdx.x & 31, ty = threadIdx.x >> 5;   // 32 x 8
#pragma unroll
  for (int i = 0; i < 32; i += 8) {
    int k = kb + ty + i, n = nb + tx;
    tile[ty + i][tx] = (k < K && n < N) ? in[(size_t)k * N + n] : 0.f;
  }
  __syncthreads();
#pragma unroll
  for (int i = 0; i < 32; i += 8) {
    int n = nb + ty + i, k = kb + tx;
    if (n < N && k < K) out[(size_t)n * K + k] = f2bf(tile[tx][ty + i]);
  }
}

// ---------------- per-row LayerNorm over C=1024 ----------------
__global__ __launch_bounds__(256) void ln_rows(const float* __restrict__ x,
                                               const float* __restrict__ g,
                                               const float* __restrict__ b,
                                               float* __restrict__ y) {
  __shared__ float s1[256], s2[256];
  const int tid = threadIdx.x;
  const size_t row = blockIdx.x;
  const float* xr = x + row * kC;
  float v0[4]; float sum = 0.f, sq = 0.f;
#pragma unroll
  for (int j = 0; j < 4; ++j) { float t = xr[j * 256 + tid]; v0[j] = t; sum += t; sq += t * t; }
  s1[tid] = sum; s2[tid] = sq; __syncthreads();
  for (int st = 128; st > 0; st >>= 1) {
    if (tid < st) { s1[tid] += s1[tid + st]; s2[tid] += s2[tid + st]; }
    __syncthreads();
  }
  const float mean = s1[0] * (1.f / kC);
  const float var  = s2[0] * (1.f / kC) - mean * mean;
  const float rstd = rsqrtf(var + 1e-5f);
  float* yr = y + row * kC;
#pragma unroll
  for (int j = 0; j < 4; ++j) { int c = j * 256 + tid; yr[c] = (v0[j] - mean) * rstd * g[c] + b[c]; }
}

// -------- token shift: dx = shift(xa)-xa ; out = bf16(xa + dx*maa) --------
__global__ __launch_bounds__(256) void shift_mix(const float* __restrict__ xa,
                                                 const float* __restrict__ maaA,
                                                 __bf16* __restrict__ outA,
                                                 const float* __restrict__ maaB,
                                                 __bf16* __restrict__ outB,
                                                 float* __restrict__ dxout) {
  const int tid = threadIdx.x;
  const size_t row = blockIdx.x;
  const int t = (int)(row % kT);
  const float* cur = xa + row * kC;
  const float* prv = cur - kC;
#pragma unroll
  for (int j = 0; j < 4; ++j) {
    int c = j * 256 + tid;
    float cv = cur[c];
    float pv = (t > 0) ? prv[c] : 0.f;
    float d = pv - cv;
    dxout[row * kC + c] = d;
    outA[row * kC + c] = f2bf(cv + d * maaA[c]);
    if (outB) outB[row * kC + c] = f2bf(cv + d * maaB[c]);
  }
}

// ---------------- templated bf16 WMMA GEMM with fused epilogues ----------------
// C = epilogue(A[M,K] @ B[K,N]); A row-major [M][K] bf16, Bt = B transposed [N][K] bf16.
// Block tile 128x128, BK=32; 8 waves in 4x2 grid, each wave = 2x4 WMMA tiles (32x64 out)
// -> 8 v_wmma_f32_16x16x32_bf16 per wave per k-step.
enum {
  EPI_NONE = 0,       // f32 out
  EPI_TANH_BF16,      // bf16 out = tanh(acc)
  EPI_SILU,           // f32 out = acc*sigmoid(acc)
  EPI_SIGMOID,        // f32 out = sigmoid(acc)
  EPI_RELU2_BF16,     // bf16 out = relu(acc)^2
  EPI_BIAS_DECAY,     // f32 out = exp(-exp(acc + aux0[n]))
  EPI_MIX_BF16,       // bf16 out = aux1[idx] + aux2[idx]*(aux0[n] + acc)
  EPI_ADD_RES         // f32 out = acc + aux2[idx]
};

#define BM 128
#define BN 128
#define BK 32
#define LDSK (BK + 8)   // 40 elems = 80B row stride; keeps 16B alignment of chunks

template <int EPI>
__global__ __launch_bounds__(256) void gemm_bf16(const __bf16* __restrict__ A, int lda,
                                                 const __bf16* __restrict__ Bt, int ldbt,
                                                 void* __restrict__ Cout, int ldc,
                                                 int M, int Nc, int K,
                                                 const float* __restrict__ aux0,
                                                 const float* __restrict__ aux1,
                                                 const float* __restrict__ aux2) {
  __shared__ __bf16 As[BM][LDSK];   // [m][k]
  __shared__ __bf16 Bs[BN][LDSK];   // [n][k]  (transposed tile)
  const int tid   = threadIdx.x;
  const int lane  = tid & 31;
  const int wave  = tid >> 5;        // 0..7
  const int waveM = wave >> 1;       // 0..3 -> 32 rows each
  const int waveN = wave & 1;        // 0..1 -> 64 cols each
  const int mBlock = blockIdx.y * BM;
  const int nBlock = blockIdx.x * BN;

  // staging: 128x32 bf16 per operand = 512 16B-chunks = 2 chunks per thread
  const int sRow0 = tid >> 2;                 // rows 0..63
  const int sRow1 = 64 + (tid >> 2);          // rows 64..127
  const int sKc   = (tid & 3) * 8;
  const bool bOk0 = (nBlock + sRow0) < Nc;
  const bool bOk1 = (nBlock + sRow1) < Nc;

  bf16x8 zb;
#pragma unroll
  for (int e = 0; e < 8; ++e) zb[e] = f2bf(0.f);

  v8f acc[2][4];
#pragma unroll
  for (int i = 0; i < 2; ++i)
#pragma unroll
    for (int j = 0; j < 4; ++j)
#pragma unroll
      for (int e = 0; e < 8; ++e) acc[i][j][e] = 0.f;

  // prologue: tile 0 into registers
  bf16x8 ra0 = *(const bf16x8*)&A[(size_t)(mBlock + sRow0) * lda + sKc];
  bf16x8 ra1 = *(const bf16x8*)&A[(size_t)(mBlock + sRow1) * lda + sKc];
  bf16x8 rb0 = bOk0 ? *(const bf16x8*)&Bt[(size_t)(nBlock + sRow0) * ldbt + sKc] : zb;
  bf16x8 rb1 = bOk1 ? *(const bf16x8*)&Bt[(size_t)(nBlock + sRow1) * ldbt + sKc] : zb;

  for (int k0 = 0; k0 < K; k0 += BK) {
    __syncthreads();   // previous tile's readers done
    *(bf16x8*)&As[sRow0][sKc] = ra0;
    *(bf16x8*)&As[sRow1][sKc] = ra1;
    *(bf16x8*)&Bs[sRow0][sKc] = rb0;
    *(bf16x8*)&Bs[sRow1][sKc] = rb1;
    __syncthreads();   // tile visible

    // issue next tile's global loads (overlap with WMMA below)
    if (k0 + BK < K) {
      ra0 = *(const bf16x8*)&A[(size_t)(mBlock + sRow0) * lda + (k0 + BK) + sKc];
      ra1 = *(const bf16x8*)&A[(size_t)(mBlock + sRow1) * lda + (k0 + BK) + sKc];
      rb0 = bOk0 ? *(const bf16x8*)&Bt[(size_t)(nBlock + sRow0) * ldbt + (k0 + BK) + sKc] : zb;
      rb1 = bOk1 ? *(const bf16x8*)&Bt[(size_t)(nBlock + sRow1) * ldbt + (k0 + BK) + sKc] : zb;
    }

    // A fragments: lanes 0-15 hold K 0-7 & 16-23, lanes 16-31 hold K 8-15 & 24-31
    const int lrow = lane & 15;
    const int hk   = (lane >> 4) * 8;
    v16bf afrag[2], bfrag[4];
#pragma unroll
    for (int ti = 0; ti < 2; ++ti) {
      const __bf16* ap = &As[waveM * 32 + ti * 16 + lrow][hk];
      bf16x8 lo = *(const bf16x8*)ap;
      bf16x8 hi = *(const bf16x8*)(ap + 16);
      afrag[ti] = (v16bf)__builtin_shufflevector(lo, hi, 0, 1, 2, 3, 4, 5, 6, 7,
                                                 8, 9, 10, 11, 12, 13, 14, 15);
    }
    // B fragments: lanes 0-15 hold K 0-7 & 16-23, lanes 16-31 hold K 8-15 & 24-31
    const int kb2 = (lane >> 4) * 8;
#pragma unroll
    for (int tj = 0; tj < 4; ++tj) {
      const __bf16* bp = &Bs[waveN * 64 + tj * 16 + lrow][kb2];
      bf16x8 lo = *(const bf16x8*)bp;
      bf16x8 hi = *(const bf16x8*)(bp + 16);
      bfrag[tj] = (v16bf)__builtin_shufflevector(lo, hi, 0, 1, 2, 3, 4, 5, 6, 7,
                                                 8, 9, 10, 11, 12, 13, 14, 15);
    }
#pragma unroll
    for (int ti = 0; ti < 2; ++ti)
#pragma unroll
      for (int tj = 0; tj < 4; ++tj)
        acc[ti][tj] = __builtin_amdgcn_wmma_f32_16x16x32_bf16(
            false, afrag[ti], false, bfrag[tj], (short)0, acc[ti][tj], false, false);
  }

  // epilogue + store (C/D layout: VGPR r -> M = r or 8+r by lane half; N = lane&15)
#pragma unroll
  for (int ti = 0; ti < 2; ++ti)
#pragma unroll
    for (int tj = 0; tj < 4; ++tj) {
      int n = nBlock + waveN * 64 + tj * 16 + (lane & 15);
      if (n >= Nc) continue;
      int mBase = mBlock + waveM * 32 + ti * 16 + ((lane >> 4) ? 8 : 0);
#pragma unroll
      for (int r = 0; r < 8; ++r) {
        int m = mBase + r;
        float val = acc[ti][tj][r];
        size_t idx = (size_t)m * ldc + n;
        if constexpr (EPI == EPI_NONE)            ((float*)Cout)[idx] = val;
        else if constexpr (EPI == EPI_TANH_BF16)  ((__bf16*)Cout)[idx] = f2bf(tanhf(val));
        else if constexpr (EPI == EPI_SILU)       ((float*)Cout)[idx] = val * sigm_(val);
        else if constexpr (EPI == EPI_SIGMOID)    ((float*)Cout)[idx] = sigm_(val);
        else if constexpr (EPI == EPI_RELU2_BF16) { float t = val > 0.f ? val : 0.f; ((__bf16*)Cout)[idx] = f2bf(t * t); }
        else if constexpr (EPI == EPI_BIAS_DECAY) { float w = val + aux0[n]; ((float*)Cout)[idx] = __expf(-__expf(w)); }
        else if constexpr (EPI == EPI_MIX_BF16)   { float xm = aux1[idx] + aux2[idx] * (aux0[n] + val); ((__bf16*)Cout)[idx] = f2bf(xm); }
        else if constexpr (EPI == EPI_ADD_RES)    ((float*)Cout)[idx] = val + aux2[idx];
      }
    }
}

// ---------------- WKV6 recurrent scan: one block per (b,h) ----------------
__global__ __launch_bounds__(256) void wkv_scan(const float* __restrict__ rp,
                                                const float* __restrict__ kp,
                                                const float* __restrict__ vp,
                                                const float* __restrict__ wp,
                                                const float* __restrict__ up,
                                                float* __restrict__ op,
                                                float* __restrict__ Sf) {
  __shared__ float rs[kN], ks[kN], vs[kN], ws_[kN], us[kN];
  __shared__ float osum[4][kN];
  const int tid = threadIdx.x;
  const int bh = blockIdx.x;            // b*kH + h
  const int b = bh / kH, h = bh % kH;
  const int m = tid & 63;
  const int gq = tid >> 6;              // 0..3
  const int n0 = gq * 16;

  float S[16];
#pragma unroll
  for (int i = 0; i < 16; ++i) S[i] = 0.f;

  if (tid < kN) us[tid] = up[h * kN + tid];
  __syncthreads();

  for (int t = 0; t < kT; ++t) {
    size_t base = (((size_t)b * kT + t) * kH + h) * kN;
    if (tid < 64)       rs[tid]        = rp[base + tid];
    else if (tid < 128) ks[tid - 64]   = kp[base + tid - 64];
    else if (tid < 192) vs[tid - 128]  = vp[base + tid - 128];
    else                ws_[tid - 192] = wp[base + tid - 192];
    __syncthreads();

    const float vm = vs[m];
    float accv = 0.f;
#pragma unroll
    for (int i = 0; i < 16; ++i) {
      int n = n0 + i;
      float kv = ks[n] * vm;
      accv += rs[n] * (S[i] + us[n] * kv);
      S[i] = ws_[n] * S[i] + kv;
    }
    osum[gq][m] = accv;
    __syncthreads();
    if (tid < 64)
      op[base + tid] = osum[0][tid] + osum[1][tid] + osum[2][tid] + osum[3][tid];
  }
#pragma unroll
  for (int i = 0; i < 16; ++i) {
    int n = n0 + i;
    Sf[(((size_t)bh) * kN + n) * kN + m] = S[i];
  }
}

// ------- per-head groupnorm (ln_x) + silu-gate, emit bf16 for Wo GEMM -------
__global__ __launch_bounds__(256) void gn_gate(const float* __restrict__ o,
                                               const float* __restrict__ g,
                                               const float* __restrict__ lng,
                                               const float* __restrict__ lnb,
                                               __bf16* __restrict__ og) {
  __shared__ float s1[16][17], s2[16][17];
  const int tid = threadIdx.x;
  const size_t row = blockIdx.x;
  const int h = tid >> 4, l = tid & 15;
  const float* orow = o + row * kC;
  float v0[4]; float sum = 0.f, sq = 0.f;
#pragma unroll
  for (int j = 0; j < 4; ++j) { float t = orow[tid * 4 + j]; v0[j] = t; sum += t; sq += t * t; }
  s1[h][l] = sum; s2[h][l] = sq; __syncthreads();
  float mu = 0.f, ss = 0.f;
#pragma unroll
  for (int j = 0; j < 16; ++j) { mu += s1[h][j]; ss += s2[h][j]; }
  mu *= (1.f / kN);
  const float var = ss * (1.f / kN) - mu * mu;
  const float rstd = rsqrtf(var + 1e-5f);
#pragma unroll
  for (int j = 0; j < 4; ++j) {
    int c = tid * 4 + j;
    float yn = (v0[j] - mu) * rstd * lng[c] + lnb[c];
    og[row * kC + c] = f2bf(yn * g[row * kC + c]);
  }
}

// ---------------- final: out = x1 + sigmoid(r)*(kk@Wv) ----------------
__global__ __launch_bounds__(256) void final_add(const float* __restrict__ x1,
                                                 const float* __restrict__ rr,
                                                 const float* __restrict__ vv,
                                                 float* __restrict__ out) {
  size_t i = (size_t)blockIdx.x * 1024 + threadIdx.x;
#pragma unroll
  for (int j = 0; j < 4; ++j, i += 256) out[i] = x1[i] + rr[i] * vv[i];
}

// ---------------- host-side dispatch ----------------
static void launch_gemm(int epi, const __bf16* A, int lda, const __bf16* Bt, int ldbt,
                        void* C, int ldc, int M, int N, int K,
                        const float* a0, const float* a1, const float* a2, hipStream_t s) {
  dim3 grid((N + BN - 1) / BN, M / BM), blk(256);
  switch (epi) {
    case EPI_NONE:       gemm_bf16<EPI_NONE><<<grid, blk, 0, s>>>(A, lda, Bt, ldbt, C, ldc, M, N, K, a0, a1, a2); break;
    case EPI_TANH_BF16:  gemm_bf16<EPI_TANH_BF16><<<grid, blk, 0, s>>>(A, lda, Bt, ldbt, C, ldc, M, N, K, a0, a1, a2); break;
    case EPI_SILU:       gemm_bf16<EPI_SILU><<<grid, blk, 0, s>>>(A, lda, Bt, ldbt, C, ldc, M, N, K, a0, a1, a2); break;
    case EPI_SIGMOID:    gemm_bf16<EPI_SIGMOID><<<grid, blk, 0, s>>>(A, lda, Bt, ldbt, C, ldc, M, N, K, a0, a1, a2); break;
    case EPI_RELU2_BF16: gemm_bf16<EPI_RELU2_BF16><<<grid, blk, 0, s>>>(A, lda, Bt, ldbt, C, ldc, M, N, K, a0, a1, a2); break;
    case EPI_BIAS_DECAY: gemm_bf16<EPI_BIAS_DECAY><<<grid, blk, 0, s>>>(A, lda, Bt, ldbt, C, ldc, M, N, K, a0, a1, a2); break;
    case EPI_MIX_BF16:   gemm_bf16<EPI_MIX_BF16><<<grid, blk, 0, s>>>(A, lda, Bt, ldbt, C, ldc, M, N, K, a0, a1, a2); break;
    case EPI_ADD_RES:    gemm_bf16<EPI_ADD_RES><<<grid, blk, 0, s>>>(A, lda, Bt, ldbt, C, ldc, M, N, K, a0, a1, a2); break;
  }
}

static void launch_cvt_t(const float* in, __bf16* out, int K, int N, hipStream_t s) {
  dim3 grid((N + 31) / 32, (K + 31) / 32), blk(256);
  cvt_t_f32_bf16<<<grid, blk, 0, s>>>(in, out, K, N);
}

extern "C" void kernel_launch(void* const* d_in, const int* in_sizes, int n_in,
                              void* d_out, int out_size, void* d_ws, size_t ws_size,
                              hipStream_t stream) {
  (void)in_sizes; (void)n_in; (void)out_size; (void)ws_size;
  const float* x         = (const float*)d_in[0];
  const float* ln0_g     = (const float*)d_in[1];
  const float* ln0_b     = (const float*)d_in[2];
  const float* ln1_g     = (const float*)d_in[3];
  const float* ln1_b     = (const float*)d_in[4];
  const float* ln2_g     = (const float*)d_in[5];
  const float* ln2_b     = (const float*)d_in[6];
  const float* lnx_g     = (const float*)d_in[7];
  const float* lnx_b     = (const float*)d_in[8];
  const float* tmaa_x    = (const float*)d_in[9];
  const float* tmaa[5]   = { (const float*)d_in[10], (const float*)d_in[11],
                             (const float*)d_in[12], (const float*)d_in[13],
                             (const float*)d_in[14] };                       // w,k,v,r,g
  const float* tm_w1     = (const float*)d_in[15];
  const float* tm_w2     = (const float*)d_in[16];
  const float* time_decay= (const float*)d_in[17];
  const float* td_w1     = (const float*)d_in[18];
  const float* td_w2     = (const float*)d_in[19];
  const float* time_faaaa= (const float*)d_in[20];
  const float* Wr        = (const float*)d_in[21];
  const float* Wk        = (const float*)d_in[22];
  const float* Wv        = (const float*)d_in[23];
  const float* Wg        = (const float*)d_in[24];
  const float* Wo        = (const float*)d_in[25];
  const float* ffn_maa_k = (const float*)d_in[26];
  const float* ffn_maa_r = (const float*)d_in[27];
  const float* Wk_ffn    = (const float*)d_in[28];
  const float* Wv_ffn    = (const float*)d_in[29];
  const float* Wr_ffn    = (const float*)d_in[30];

  // bump allocator over d_ws
  size_t off = 0;
  auto alloc = [&](size_t elems, size_t esz) -> void* {
    void* p = (char*)d_ws + off;
    off += (elems * esz + 255) & ~(size_t)255;
    return p;
  };

  // transposed bf16 weight shadows  (Bt layout: [N][K])
  __bf16* tm_w1_t  = (__bf16*)alloc((size_t)5 * kTM * kC, 2);   // [160][1024]
  __bf16* tm_w2_t  = (__bf16*)alloc((size_t)5 * kC * kTM, 2);   // 5 x [1024][32]
  __bf16* td_w1_t  = (__bf16*)alloc((size_t)kTD * kC, 2);       // [64][1024]
  __bf16* td_w2_t  = (__bf16*)alloc((size_t)kC * kTD, 2);       // [1024][64]
  __bf16* Wr_t     = (__bf16*)alloc((size_t)kC * kC, 2);
  __bf16* Wk_t     = (__bf16*)alloc((size_t)kC * kC, 2);
  __bf16* Wv_t     = (__bf16*)alloc((size_t)kC * kC, 2);
  __bf16* Wg_t     = (__bf16*)alloc((size_t)kC * kC, 2);
  __bf16* Wo_t     = (__bf16*)alloc((size_t)kC * kC, 2);
  __bf16* Wkffn_t  = (__bf16*)alloc((size_t)kF * kC, 2);        // [3584][1024]
  __bf16* Wvffn_t  = (__bf16*)alloc((size_t)kC * kF, 2);        // [1024][3584]
  __bf16* Wrffn_t  = (__bf16*)alloc((size_t)kC * kC, 2);

  // activations
  float*  x0       = (float*)alloc(kBTC, 4);
  float*  xa       = (float*)alloc(kBTC, 4);
  float*  dx       = (float*)alloc(kBTC, 4);
  __bf16* xxx_bf   = (__bf16*)alloc(kBTC, 2);
  __bf16* lora_bf  = (__bf16*)alloc((size_t)kBT * 5 * kTM, 2);
  __bf16* xmix[5];
  for (int f = 0; f < 5; ++f) xmix[f] = (__bf16*)alloc(kBTC, 2);
  float*  r_f      = (float*)alloc(kBTC, 4);
  float*  k_f      = (float*)alloc(kBTC, 4);
  float*  v_f      = (float*)alloc(kBTC, 4);
  float*  decay_f  = (float*)alloc(kBTC, 4);
  __bf16* wtan_bf  = (__bf16*)alloc((size_t)kBT * kTD, 2);
  float*  g_f      = (float*)alloc(kBTC, 4);
  float*  o_f      = (float*)alloc(kBTC, 4);
  __bf16* og_bf    = (__bf16*)alloc(kBTC, 2);
  float*  x1       = (float*)alloc(kBTC, 4);
  __bf16* kk_bf    = (__bf16*)alloc((size_t)kBT * kF, 2);

  // lifetime-based aliases
  float*  xf   = xa;        // ln2 output (xa dead after mix GEMMs)
  float*  dxf  = dx;        // ffn token-shift delta
  __bf16* xkin = xmix[0];   // ffn k input (xw dead after td_w1 GEMM)
  __bf16* xrin = xmix[1];   // ffn r input
  float*  rr_f = r_f;       // sigmoid(r) (r dead after scan)
  float*  vv_f = k_f;       // kk@Wv_ffn

  float* out_x  = (float*)d_out;
  float* out_Sf = (float*)d_out + kBTC;

  // ---- 0. weight conversion + transpose (B operands for all GEMMs) ----
  launch_cvt_t(tm_w1, tm_w1_t, kC, 5 * kTM, stream);
  for (int f = 0; f < 5; ++f)
    launch_cvt_t(tm_w2 + (size_t)f * kTM * kC, tm_w2_t + (size_t)f * kC * kTM, kTM, kC, stream);
  launch_cvt_t(td_w1, td_w1_t, kC, kTD, stream);
  launch_cvt_t(td_w2, td_w2_t, kTD, kC, stream);
  launch_cvt_t(Wr, Wr_t, kC, kC, stream);
  launch_cvt_t(Wk, Wk_t, kC, kC, stream);
  launch_cvt_t(Wv, Wv_t, kC, kC, stream);
  launch_cvt_t(Wg, Wg_t, kC, kC, stream);
  launch_cvt_t(Wo, Wo_t, kC, kC, stream);
  launch_cvt_t(Wk_ffn, Wkffn_t, kC, kF, stream);
  launch_cvt_t(Wv_ffn, Wvffn_t, kF, kC, stream);
  launch_cvt_t(Wr_ffn, Wrffn_t, kC, kC, stream);

  // ---- 1. ln0, ln1, token shift + maa_x mix ----
  ln_rows<<<kBT, 256, 0, stream>>>(x, ln0_g, ln0_b, x0);
  ln_rows<<<kBT, 256, 0, stream>>>(x0, ln1_g, ln1_b, xa);
  shift_mix<<<kBT, 256, 0, stream>>>(xa, tmaa_x, xxx_bf, nullptr, nullptr, dx);

  // ---- 2. LoRA token-shift mixes ----
  launch_gemm(EPI_TANH_BF16, xxx_bf, kC, tm_w1_t, kC, lora_bf, 5 * kTM,
              kBT, 5 * kTM, kC, nullptr, nullptr, nullptr, stream);
  for (int f = 0; f < 5; ++f)
    launch_gemm(EPI_MIX_BF16, lora_bf + f * kTM, 5 * kTM, tm_w2_t + (size_t)f * kC * kTM, kTM,
                xmix[f], kC, kBT, kC, kTM, tmaa[f], xa, dx, stream);

  // ---- 3. projections ----
  launch_gemm(EPI_NONE, xmix[3], kC, Wr_t, kC, r_f, kC, kBT, kC, kC, nullptr, nullptr, nullptr, stream); // r
  launch_gemm(EPI_NONE, xmix[1], kC, Wk_t, kC, k_f, kC, kBT, kC, kC, nullptr, nullptr, nullptr, stream); // k
  launch_gemm(EPI_NONE, xmix[2], kC, Wv_t, kC, v_f, kC, kBT, kC, kC, nullptr, nullptr, nullptr, stream); // v
  launch_gemm(EPI_SILU, xmix[4], kC, Wg_t, kC, g_f, kC, kBT, kC, kC, nullptr, nullptr, nullptr, stream); // g
  launch_gemm(EPI_TANH_BF16, xmix[0], kC, td_w1_t, kC, wtan_bf, kTD,
              kBT, kTD, kC, nullptr, nullptr, nullptr, stream);                                          // tanh(xw@td_w1)
  launch_gemm(EPI_BIAS_DECAY, wtan_bf, kTD, td_w2_t, kTD, decay_f, kC,
              kBT, kC, kTD, time_decay, nullptr, nullptr, stream);                                       // exp(-exp(w))

  // ---- 4. WKV scan (writes o and Sf) ----
  wkv_scan<<<kB * kH, 256, 0, stream>>>(r_f, k_f, v_f, decay_f, time_faaaa, o_f, out_Sf);

  // ---- 5. per-head groupnorm + gate, output projection + residual ----
  gn_gate<<<kBT, 256, 0, stream>>>(o_f, g_f, lnx_g, lnx_b, og_bf);
  launch_gemm(EPI_ADD_RES, og_bf, kC, Wo_t, kC, x1, kC, kBT, kC, kC, nullptr, nullptr, x0, stream);

  // ---- 6. channel mix (FFN) ----
  ln_rows<<<kBT, 256, 0, stream>>>(x1, ln2_g, ln2_b, xf);
  shift_mix<<<kBT, 256, 0, stream>>>(xf, ffn_maa_k, xkin, ffn_maa_r, xrin, dxf);
  launch_gemm(EPI_RELU2_BF16, xkin, kC, Wkffn_t, kC, kk_bf, kF, kBT, kF, kC, nullptr, nullptr, nullptr, stream);
  launch_gemm(EPI_SIGMOID, xrin, kC, Wrffn_t, kC, rr_f, kC, kBT, kC, kC, nullptr, nullptr, nullptr, stream);
  launch_gemm(EPI_NONE, kk_bf, kF, Wvffn_t, kF, vv_f, kC, kBT, kC, kF, nullptr, nullptr, nullptr, stream);

  // ---- 7. final combine ----
  final_add<<<kBT * kC / 1024, 256, 0, stream>>>(x1, rr_f, vv_f, out_x);
}
